// EdgeConv_3934190044183
// MI455X (gfx1250) — matrix operand order repaired and checked by reading, not compiled, not run on hardware
//
#include <hip/hip_runtime.h>
#include <hip/hip_bf16.h>
#include <math.h>

// ---------------------------------------------------------------------------
// EdgeConv (gather -> [y-x, x] -> 1x1 conv GEMM -> InstanceNorm -> LeakyReLU
//          -> mean over K) for MI455X / gfx1250, wave32 + WMMA bf16.
//
// Shapes: B=2, N=1024, C=32, D=4 (P=64), K=8, O=64, 2C=64.
// Split GEMM:  h = W[:,0:32] @ (y - x)  +  W[:,32:64] @ x
// The second term is k-invariant -> computed once per point and used as the
// WMMA C-input accumulator for every neighbor tile.
// ---------------------------------------------------------------------------

typedef __attribute__((ext_vector_type(16))) __bf16 v16bf;
typedef __attribute__((ext_vector_type(8)))  __bf16 v8bf;
typedef __attribute__((ext_vector_type(8)))  float  v8f;

#define NPT   1024      // points per batch
#define CCH   32        // input channels
#define PP    64        // spatial (4^3)
#define KK    8         // neighbors
#define OO    64        // output channels
#define JW    512       // KK*PP columns of the y-GEMM per point
#define SY_STR 40       // LDS row stride (bf16) for [j][c] arrays (20 banks, conflict-free)
#define SW_STR 72       // LDS row stride (bf16) for W [o][c] (36 banks, conflict-free)
#define NKP_F 524288.0f // N*K*P elements per (b,o) for InstanceNorm
#define EPS   1e-5f
#define NEG_SLOPE 0.01f

// ---- fragment loaders (ISA 7.12.2 layouts, wave32) -------------------------

// A fragment: 16x32 bf16, row M = o_t*16 + (lane%16).
// elems 0..7  -> K = c0 + 8*half + e ; elems 8..15 -> K = c0 + 16 + 8*half + e
__device__ __forceinline__ v16bf load_a_frag(const __bf16* sW, int o_t, int ln, int half, int c0) {
    const __bf16* base = sW + (o_t * 16 + ln) * SW_STR + c0 + 8 * half;
    union { v16bf v; v8bf h[2]; } u;
    u.h[0] = *(const v8bf*)(base);
    u.h[1] = *(const v8bf*)(base + 16);
    return u.v;
}

// B fragment: 32x16 bf16 from [j][c] LDS array; col N = j-tile base + lane%16.
// elems 0..7 -> K = 8*half + e ; elems 8..15 -> K = 16 + 8*half + e
__device__ __forceinline__ v16bf load_b_frag(const __bf16* arr, int j, int half) {
    const __bf16* base = arr + j * SY_STR + 8 * half;
    union { v16bf v; v8bf h[2]; } u;
    u.h[0] = *(const v8bf*)(base);
    u.h[1] = *(const v8bf*)(base + 16);
    return u.v;
}

// ---- shared staging: W, center x, y-x, x^T ---------------------------------
__device__ __forceinline__ void stage_inputs(const float* __restrict__ x,
                                             const int*   __restrict__ ind,
                                             const float* __restrict__ W,
                                             __bf16* sW, __bf16* sY, __bf16* sX,
                                             float* sxc, int g, int t) {
    // W[o][c] -> bf16, padded stride
    for (int i = t; i < OO * 2 * CCH; i += 256) {
        int o = i >> 6, c = i & 63;
        sW[o * SW_STR + c] = (__bf16)W[i];
    }
    // center x [c][p] as float (reused for all 8 subtractions)
    const float* xc = x + (size_t)g * (CCH * PP);
    for (int i = t; i < CCH * PP; i += 256) sxc[i] = xc[i];
    __syncthreads();

    // x^T -> sX[p][c] bf16 (B matrix of the k-invariant GEMM)
    for (int i = t; i < CCH * PP; i += 256) {
        int c = i >> 6, p = i & 63;
        sX[p * SY_STR + c] = (__bf16)sxc[i];
    }
    // gathered neighbors: sY[j = k*64+p][c] = bf16(y - x)
    for (int k = 0; k < KK; ++k) {
        const int nb = ind[g * KK + k];
        const float* yp = x + (size_t)nb * (CCH * PP);
        for (int i = t; i < CCH * PP; i += 256) {
            int c = i >> 6, p = i & 63;
            sY[(k * PP + p) * SY_STR + c] = (__bf16)(yp[i] - sxc[i]);
        }
    }
    __syncthreads();
}

// ---- pass 0: zero the global accumulators ----------------------------------
__global__ void k_init(float* ws) {
    ws[threadIdx.x] = 0.0f;   // gsum[128] ++ gsumsq[128]
}

// ---- pass 1: GEMM + global sum / sum-of-squares per (b, o) -----------------
__global__ __launch_bounds__(256) void k_pass1(const float* __restrict__ x,
                                               const int*   __restrict__ ind,
                                               const float* __restrict__ W,
                                               float* __restrict__ gsum,
                                               float* __restrict__ gsumsq) {
    __shared__ alignas(16) __bf16 sW[OO * SW_STR];
    __shared__ alignas(16) __bf16 sY[JW * SY_STR];
    __shared__ alignas(16) __bf16 sX[PP * SY_STR];
    __shared__ float sxc[CCH * PP];
    __shared__ float ssum[OO], ssq[OO];

    const int t = threadIdx.x;
    const int g = blockIdx.x;          // b*N + n
    const int b = g >> 10;

    if (t < OO) { ssum[t] = 0.0f; ssq[t] = 0.0f; }
    stage_inputs(x, ind, W, sW, sY, sX, sxc, g, t);

    const int w = t >> 5, lane = t & 31;
    const int half = lane >> 4, ln = lane & 15;
    const int o_t = w >> 1, pt0 = (w & 1) * 2;

    const v16bf a_y = load_a_frag(sW, o_t, ln, half, 0);    // W[:, 0:32]
    const v16bf a_x = load_a_frag(sW, o_t, ln, half, 32);   // W[:, 32:64]

    float s1[8], s2[8];
#pragma unroll
    for (int r = 0; r < 8; ++r) { s1[r] = 0.0f; s2[r] = 0.0f; }

    for (int pi = 0; pi < 2; ++pi) {
        const int pt = pt0 + pi;
        // k-invariant part: hx = W_x @ x  (one WMMA, zero C)
        v8f hx = {};
        const v16bf bx = load_b_frag(sX, pt * 16 + ln, half);
        hx = __builtin_amdgcn_wmma_f32_16x16x32_bf16(false, a_x, false, bx,
                                                     (short)0, hx, false, false);
#pragma unroll
        for (int k = 0; k < KK; ++k) {
            const v16bf by = load_b_frag(sY, k * PP + pt * 16 + ln, half);
            v8f acc = __builtin_amdgcn_wmma_f32_16x16x32_bf16(false, a_y, false, by,
                                                              (short)0, hx, false, false);
#pragma unroll
            for (int r = 0; r < 8; ++r) {
                const float v = acc[r];
                s1[r] += v;
                s2[r] += v * v;
            }
        }
    }
    // per-row (o) reduction: lane owns rows o_t*16 + 8*half + r
#pragma unroll
    for (int r = 0; r < 8; ++r) {
        const int o = o_t * 16 + 8 * half + r;
        atomicAdd(&ssum[o], s1[r]);
        atomicAdd(&ssq[o],  s2[r]);
    }
    __syncthreads();
    if (t < OO) {
        atomicAdd(&gsum[b * OO + t],   ssum[t]);
        atomicAdd(&gsumsq[b * OO + t], ssq[t]);
    }
}

// ---- pass 2: mean / rsqrt(var+eps) per (b, o) ------------------------------
__global__ void k_finalize(const float* gsum, const float* gsumsq,
                           float* gmean, float* grstd) {
    const int i = threadIdx.x;   // 128 = B*O
    const float m = gsum[i] / NKP_F;
    const float v = gsumsq[i] / NKP_F - m * m;
    gmean[i] = m;
    grstd[i] = rsqrtf(v + EPS);
}

// ---- pass 3: recompute GEMM, normalize, LeakyReLU, mean over k, store ------
__global__ __launch_bounds__(256) void k_pass3(const float* __restrict__ x,
                                               const int*   __restrict__ ind,
                                               const float* __restrict__ W,
                                               const float* __restrict__ gmean,
                                               const float* __restrict__ grstd,
                                               float* __restrict__ out) {
    __shared__ alignas(16) __bf16 sW[OO * SW_STR];
    __shared__ alignas(16) __bf16 sY[JW * SY_STR];
    __shared__ alignas(16) __bf16 sX[PP * SY_STR];
    __shared__ float sxc[CCH * PP];

    const int t = threadIdx.x;
    const int g = blockIdx.x;          // b*N + n
    const int b = g >> 10;

    stage_inputs(x, ind, W, sW, sY, sX, sxc, g, t);

    const int w = t >> 5, lane = t & 31;
    const int half = lane >> 4, ln = lane & 15;
    const int o_t = w >> 1, pt0 = (w & 1) * 2;

    const v16bf a_y = load_a_frag(sW, o_t, ln, half, 0);
    const v16bf a_x = load_a_frag(sW, o_t, ln, half, 32);

    float mr[8], sr[8];
#pragma unroll
    for (int r = 0; r < 8; ++r) {
        const int o = o_t * 16 + 8 * half + r;
        mr[r] = gmean[b * OO + o];
        sr[r] = grstd[b * OO + o];
    }

    for (int pi = 0; pi < 2; ++pi) {
        const int pt = pt0 + pi;
        v8f hx = {};
        const v16bf bx = load_b_frag(sX, pt * 16 + ln, half);
        hx = __builtin_amdgcn_wmma_f32_16x16x32_bf16(false, a_x, false, bx,
                                                     (short)0, hx, false, false);
        float oacc[8];
#pragma unroll
        for (int r = 0; r < 8; ++r) oacc[r] = 0.0f;

#pragma unroll
        for (int k = 0; k < KK; ++k) {
            const v16bf by = load_b_frag(sY, k * PP + pt * 16 + ln, half);
            v8f acc = __builtin_amdgcn_wmma_f32_16x16x32_bf16(false, a_y, false, by,
                                                              (short)0, hx, false, false);
#pragma unroll
            for (int r = 0; r < 8; ++r) {
                float v = (acc[r] - mr[r]) * sr[r];
                v = (v >= 0.0f) ? v : NEG_SLOPE * v;
                oacc[r] += v;
            }
        }
        // out[b, n, o, p] : flat ((g*OO + o)*PP + p)
        const int p = pt * 16 + ln;
#pragma unroll
        for (int r = 0; r < 8; ++r) {
            const int o = o_t * 16 + 8 * half + r;
            out[((size_t)g * OO + o) * PP + p] = oacc[r] * (1.0f / (float)KK);
        }
    }
}

// ---------------------------------------------------------------------------
extern "C" void kernel_launch(void* const* d_in, const int* in_sizes, int n_in,
                              void* d_out, int out_size, void* d_ws, size_t ws_size,
                              hipStream_t stream) {
    const float* x   = (const float*)d_in[0];   // [B, N, C, 4,4,4] f32
    const int*   ind = (const int*)d_in[1];     // [B, N, K] int
    const float* W   = (const float*)d_in[2];   // [O, 2C] f32
    float* out = (float*)d_out;                 // [B, N, O, 4,4,4] f32
    float* ws  = (float*)d_ws;

    float* gsum   = ws;            // [128]
    float* gsumsq = ws + 128;      // [128]
    float* gmean  = ws + 256;      // [128]
    float* grstd  = ws + 384;      // [128]

    const int nblocks = 2 * NPT;   // one workgroup per point (b, n)

    k_init<<<1, 256, 0, stream>>>(ws);
    k_pass1<<<nblocks, 256, 0, stream>>>(x, ind, W, gsum, gsumsq);
    k_finalize<<<1, 128, 0, stream>>>(gsum, gsumsq, gmean, grstd);
    k_pass3<<<nblocks, 256, 0, stream>>>(x, ind, W, gmean, grstd, out);
}
